// TransFusionHead_42528766165974
// MI455X (gfx1250) — matrix operand order repaired and checked by reading, not compiled, not run on hardware
//
#include <hip/hip_runtime.h>
#include <math.h>

#define DEVI __device__ __forceinline__

typedef __attribute__((ext_vector_type(16))) _Float16 v16h;
typedef __attribute__((ext_vector_type(8)))  _Float16 v8h;
typedef __attribute__((ext_vector_type(8)))  float    v8f;

#define BB 2
#define CIN0 512
#define HH 180
#define HW 32400          // 180*180
#define HID 128
#define NCLS 10
#define NPROP 200
#define NPR_T 400         // B*NPROP
#define RADW 41
#define WIN2 1681
#define FFN_D 256
#define EPS 1e-5f
#define NPIX (BB*HW)      // 64800

// ---------------------------------------------------------------- WMMA utils
DEVI v8f wmma32(v16h a, v16h b, v8f c) {
  return __builtin_amdgcn_wmma_f32_16x16x32_f16(false, a, false, b, (short)0, c, false, false);
}

// A fragment: lane group g (=lane>>4): halves 0..15 = K[g*16..g*16+15]
DEVI v16h afrag(const _Float16* row, int g) {
  v8h a0 = *(const v8h*)(row + g * 16);
  v8h a1 = *(const v8h*)(row + g * 16 + 8);
  v16h r;
#pragma unroll
  for (int i = 0; i < 8; ++i) { r[i] = a0[i]; r[i + 8] = a1[i]; }
  return r;
}

// ---------------------------------------------------------------- small prep kernels
__global__ __launch_bounds__(256) void k_convert_x(const float* __restrict__ xin,
                                                   _Float16* __restrict__ xh, int n) {
  int e = blockIdx.x * 256 + threadIdx.x;
  if (e >= n) return;
  int ci = e % CIN0;
  int p  = e / CIN0;
  int x  = p % HH;
  int y  = (p / HH) % HH;
  int b  = p / (HH * HH);
  xh[e] = (_Float16)xin[(((size_t)(b * CIN0 + ci) * HH) + y) * HH + x];
}

// pack OIHW conv weight -> Bt[o][(ky*3+kx)*Cin + i], zero pad rows o>=Cout
__global__ __launch_bounds__(256) void k_pack_conv(const float* __restrict__ src,
                                                   _Float16* __restrict__ dst,
                                                   int Cout, int Cin, int CoutPad, int n) {
  int e = blockIdx.x * 256 + threadIdx.x;
  if (e >= n) return;
  int i   = e % Cin;
  int kk9 = (e / Cin) % 9;
  int o   = e / (9 * Cin);
  float v = 0.f;
  if (o < Cout) {
    int ky = kk9 / 3, kx = kk9 % 3;
    v = src[(((size_t)(o * Cin + i) * 3) + ky) * 3 + kx];
  }
  dst[e] = (_Float16)v;
}

__global__ __launch_bounds__(256) void k_half_copy(const float* __restrict__ src,
                                                   _Float16* __restrict__ dst, int n) {
  int e = blockIdx.x * 256 + threadIdx.x;
  if (e < n) dst[e] = (_Float16)src[e];
}

// ---------------------------------------------------------------- implicit-GEMM 3x3 conv (WMMA)
// MODE 0: out = conv+bias  -> f16 NHWC
// MODE 1: out = relu(bn(conv+bias)) -> f16 NHWC
// MODE 2: COUTP=16 (10 real): dense heatmap f32 NCHW + sigmoid map f32 NCHW
template <int CIN_, int COUTP, int MODE>
__global__ __launch_bounds__(256) void k_conv(
    const _Float16* __restrict__ xin, const _Float16* __restrict__ wp,
    const float* __restrict__ bias,
    const float* __restrict__ bng, const float* __restrict__ bnb,
    const float* __restrict__ bnm, const float* __restrict__ bnv,
    _Float16* __restrict__ outh, float* __restrict__ outf, float* __restrict__ outsig) {
  constexpr int NT = COUTP / 16;
  __shared__ _Float16 lA[128 * 48];
  __shared__ _Float16 lB[COUTP * 48];
  __shared__ float sScale[128];
  __shared__ float sShift[128];

  const int t = threadIdx.x, wave = t >> 5, lane = t & 31, g = lane >> 4, nl = lane & 15;
  const int m0 = blockIdx.x * 128;

  if (MODE == 1) {
    if (t < COUTP) {
      float sc = bng[t] * rsqrtf(bnv[t] + EPS);
      sScale[t] = sc;
      sShift[t] = bnb[t] - bnm[t] * sc + bias[t] * sc;
    }
    __syncthreads();
  }

  v8f acc[NT];
#pragma unroll
  for (int jn = 0; jn < NT; ++jn)
#pragma unroll
    for (int r = 0; r < 8; ++r) acc[jn][r] = 0.f;

  for (int kk9 = 0; kk9 < 9; ++kk9) {
    const int dy = kk9 / 3 - 1, dx = kk9 % 3 - 1;
    for (int ci0 = 0; ci0 < CIN_; ci0 += 32) {
      { // A tile: 128 pixels x 32 ch (f16)
        int r = t >> 1, hofs = (t & 1) * 16;
        int m = m0 + r;
        v8h z0, z1;
        bool ok = false;
        const _Float16* src = nullptr;
        if (m < NPIX) {
          int b = m / HW, yx = m % HW, y = yx / HH, x = yx % HH;
          int iy = y + dy, ix = x + dx;
          if (iy >= 0 && iy < HH && ix >= 0 && ix < HH) {
            ok = true;
            src = xin + ((size_t)((b * HH + iy) * HH + ix)) * CIN_ + ci0 + hofs;
          }
        }
        if (ok) { z0 = *(const v8h*)src; z1 = *(const v8h*)(src + 8); }
        else {
#pragma unroll
          for (int i = 0; i < 8; ++i) { z0[i] = (_Float16)0.f; z1[i] = (_Float16)0.f; }
        }
        *(v8h*)&lA[r * 48 + hofs]     = z0;
        *(v8h*)&lA[r * 48 + hofs + 8] = z1;
      }
      { // B tile: COUTP rows x 32 ch
        int o = t >> 1, hofs = (t & 1) * 16;
        if (o < COUTP) {
          const _Float16* src = wp + (size_t)o * (9 * CIN_) + kk9 * CIN_ + ci0 + hofs;
          *(v8h*)&lB[o * 48 + hofs]     = *(const v8h*)src;
          *(v8h*)&lB[o * 48 + hofs + 8] = *(const v8h*)(src + 8);
        }
      }
      __syncthreads();
      v16h af = afrag(&lA[(wave * 16 + nl) * 48], g);
#pragma unroll
      for (int jn = 0; jn < NT; ++jn) {
        v16h bf = *(const v16h*)&lB[(jn * 16 + nl) * 48 + g * 16];
        acc[jn] = wmma32(af, bf, acc[jn]);
      }
      __syncthreads();
    }
  }

#pragma unroll
  for (int jn = 0; jn < NT; ++jn) {
#pragma unroll
    for (int r = 0; r < 8; ++r) {
      int m = m0 + wave * 16 + g * 8 + r;
      if (m < NPIX) {
        int n = jn * 16 + nl;
        float v = acc[jn][r];
        if (MODE == 0) {
          outh[(size_t)m * HID + n] = (_Float16)(v + bias[n]);
        } else if (MODE == 1) {
          float y = v * sScale[n] + sShift[n];
          outh[(size_t)m * HID + n] = (_Float16)fmaxf(y, 0.f);
        } else {
          if (n < NCLS) {
            float y = v + bias[n];
            int b = m / HW, yx = m % HW;
            size_t o = (size_t)(b * NCLS + n) * HW + yx;
            outf[o]   = y;
            outsig[o] = 1.f / (1.f + expf(-y));
          }
        }
      }
    }
  }
}

// ---------------------------------------------------------------- NMS (local max)
__global__ __launch_bounds__(256) void k_nms(const float* __restrict__ hs,
                                             float* __restrict__ out, int n) {
  int i = blockIdx.x * 256 + threadIdx.x;
  if (i >= n) return;
  int yx = i % HW, bc = i / HW, c = bc % NCLS;
  float h = hs[i];
  float keep;
  if (c >= 8) keep = h;  // nuScenes k=1 classes
  else {
    int y = yx / HH, x = yx % HH;
    if (y == 0 || y == HH - 1 || x == 0 || x == HH - 1) keep = 0.f;
    else {
      float m = -3.0e38f;
#pragma unroll
      for (int dy = -1; dy <= 1; ++dy)
#pragma unroll
        for (int dx = -1; dx <= 1; ++dx) m = fmaxf(m, hs[i + dy * HH + dx]);
      keep = (h == m) ? h : 0.f;
    }
  }
  out[i] = keep;
}

// ---------------------------------------------------------------- exact top-k (argmax iterate, tie->lower idx)
__global__ __launch_bounds__(1024) void k_topk(float* __restrict__ hm,
                                               int* __restrict__ topcls,
                                               int* __restrict__ toppix) {
  __shared__ float sv[1024];
  __shared__ int   si[1024];
  const int b = blockIdx.x, t = threadIdx.x;
  float* base = hm + (size_t)b * NCLS * HW;
  for (int it = 0; it < NPROP; ++it) {
    float bv = -3.0e38f;
    int bi = 0x7fffffff;
    for (int i = t; i < NCLS * HW; i += 1024) {
      float v = base[i];
      if (v > bv || (v == bv && i < bi)) { bv = v; bi = i; }
    }
    sv[t] = bv; si[t] = bi;
    __syncthreads();
    for (int st = 512; st > 0; st >>= 1) {
      if (t < st) {
        float v2 = sv[t + st]; int i2 = si[t + st];
        if (v2 > sv[t] || (v2 == sv[t] && i2 < si[t])) { sv[t] = v2; si[t] = i2; }
      }
      __syncthreads();
    }
    if (t == 0) {
      int idx = si[0];
      base[idx] = -1e30f;
      topcls[b * NPROP + it] = idx / HW;
      toppix[b * NPROP + it] = idx % HW;
    }
    __syncthreads();
  }
}

// ---------------------------------------------------------------- per-proposal prep (q + self-attn + Qcross)
__global__ __launch_bounds__(128) void k_prep(
    const _Float16* __restrict__ feath, const float* cew, const float* ceb,
    const float* sw1, const float* sb1, const float* sbng, const float* sbnb,
    const float* sbnm, const float* sbnv, const float* sw2, const float* sb2,
    const float* wv_s, const float* bv_s, const float* wo_s, const float* bo_s,
    const float* n1g, const float* n1b, const float* wq_c, const float* bq_c,
    const int* topcls, const int* toppix, float* q1ws, float* qcws) {
  __shared__ float bufA[128], bufB[128], red[128];
  const int j = blockIdx.x, c = threadIdx.x;
  const int b = j / NPROP;
  const int pix = toppix[j], cls = topcls[j];
  const float qx = (float)(pix / HH) + 0.5f, qy = (float)(pix % HH) + 0.5f;

  float rq = (float)feath[(size_t)(b * HW + pix) * HID + c] + cew[c * NCLS + cls] + ceb[c];
  // self posembed
  float s = sw1[c * 2] * qx + sw1[c * 2 + 1] * qy + sb1[c];
  float sc = sbng[c] * rsqrtf(sbnv[c] + EPS);
  s = (s - sbnm[c]) * sc + sbnb[c];
  bufA[c] = fmaxf(s, 0.f);
  __syncthreads();
  float rqpe = sb2[c];
  for (int k = 0; k < 128; ++k) rqpe += sw2[c * 128 + k] * bufA[k];
  __syncthreads();
  bufA[c] = rq + rqpe;  // qq
  __syncthreads();
  float vsa = bv_s[c];
  for (int k = 0; k < 128; ++k) vsa += wv_s[c * 128 + k] * bufA[k];
  bufB[c] = vsa;
  __syncthreads();
  float sao = bo_s[c];
  for (int k = 0; k < 128; ++k) sao += wo_s[c * 128 + k] * bufB[k];
  float tq = rq + sao;
  // LN1
  red[c] = tq; __syncthreads();
  for (int st = 64; st > 0; st >>= 1) { if (c < st) red[c] += red[c + st]; __syncthreads(); }
  float mu = red[0] * (1.f / 128.f); __syncthreads();
  float d = tq - mu;
  red[c] = d * d; __syncthreads();
  for (int st = 64; st > 0; st >>= 1) { if (c < st) red[c] += red[c + st]; __syncthreads(); }
  float var = red[0] * (1.f / 128.f); __syncthreads();
  float q1 = d * rsqrtf(var + EPS) * n1g[c] + n1b[c];
  q1ws[(size_t)j * 128 + c] = q1;
  bufB[c] = q1 + rqpe;
  __syncthreads();
  float qc = bq_c[c];
  for (int k = 0; k < 128; ++k) qc += wq_c[c * 128 + k] * bufB[k];
  qcws[(size_t)j * 128 + c] = qc;
}

// ---------------------------------------------------------------- fused cross-attention decoder + heads
struct HeadP {
  const float *w1, *b1, *g, *be, *mm, *vv, *w2, *b2;
  int nout, cb;
};
struct CrossArgs {
  const _Float16 *feath, *pew2, *wkh, *wvh;
  const float *pe_w1, *pe_b1, *pe_g, *pe_b, *pe_m, *pe_v, *pe_b2;
  const float *bk, *bv, *wo, *bo;
  const float *n2g, *n2b, *n3g, *n3b;
  const float *l1w, *l1b, *l2w, *l2b;
  const int* toppix;
  const float *q1ws, *qcws;
  HeadP heads[6];
  float* preds;
};

DEVI void gemm128(const _Float16* Abuf, const _Float16* __restrict__ Bglob,
                  _Float16* sBp, v8f* acc, int t) {
  const int wave = t >> 5, lane = t & 31, g = lane >> 4, nl = lane & 15;
  for (int k0 = 0; k0 < 128; k0 += 32) {
    {
      int o = t >> 1, hofs = (t & 1) * 16;
      const _Float16* src = Bglob + (size_t)o * 128 + k0 + hofs;
      *(v8h*)&sBp[o * 48 + hofs]     = *(const v8h*)src;
      *(v8h*)&sBp[o * 48 + hofs + 8] = *(const v8h*)(src + 8);
      if (k0 + 32 < 128) __builtin_prefetch(src + 32, 0, 3);
    }
    __syncthreads();
    v16h af = afrag(Abuf + (wave * 16 + nl) * 144 + k0, g);
#pragma unroll
    for (int jn = 0; jn < 8; ++jn) {
      v16h bf = *(const v16h*)&sBp[(jn * 16 + nl) * 48 + g * 16];
      acc[jn] = wmma32(af, bf, acc[jn]);
    }
    __syncthreads();
  }
}

DEVI float ln256(float v, int t, float* red, const float* gg, const float* bb2) {
  red[t] = (t < 128) ? v : 0.f; __syncthreads();
#pragma unroll
  for (int st = 128; st > 0; st >>= 1) { if (t < st) red[t] += red[t + st]; __syncthreads(); }
  float mu = red[0] * (1.f / 128.f); __syncthreads();
  float d = v - mu;
  red[t] = (t < 128) ? d * d : 0.f; __syncthreads();
#pragma unroll
  for (int st = 128; st > 0; st >>= 1) { if (t < st) red[t] += red[t + st]; __syncthreads(); }
  float var = red[0] * (1.f / 128.f); __syncthreads();
  return (t < 128) ? (d * rsqrtf(var + EPS) * gg[t] + bb2[t]) : 0.f;
}

__global__ __launch_bounds__(256) void k_cross(CrossArgs A) {
  __shared__ _Float16 sH[128 * 144];    // h, then Khat, then V staging
  __shared__ _Float16 sKK[128 * 144];   // kk = kf + kpe
  __shared__ _Float16 sB[128 * 48];
  __shared__ float sS[128 * 8];
  __shared__ float sO[128];
  __shared__ float sM[8], sL[8], sAl[8];
  __shared__ int sKI[128], sKMsk[128];
  __shared__ float sKX[128], sKY[128];
  __shared__ float sRed[256];
  __shared__ float sVa[128];
  __shared__ float sVb[256];
  __shared__ float sQ[128];

  const int j = blockIdx.x, b = j / NPROP, jj = j % NPROP;
  const int t = threadIdx.x, wave = t >> 5, lane = t & 31, g = lane >> 4, nl = lane & 15;
  const int pix = A.toppix[j];
  const int px = pix / HH, py = pix % HH;
  const float NEGINF = -__builtin_inff();

  if (t < 128) { sQ[t] = A.qcws[(size_t)j * 128 + t]; sO[t] = 0.f; }
  if (t < 8) { sM[t] = NEGINF; sL[t] = 0.f; }
  __syncthreads();

  for (int tile = 0; tile < 14; ++tile) {
    const int t0 = tile * 128;
    if (t < 128) {
      int key = t0 + t;
      int rx = key / RADW - 20, ry = key % RADW - 20;
      int kidx = (px + rx) * HH + (py + ry);
      int msk = (key >= WIN2) || (kidx < 0) || (kidx >= HW);
      kidx = kidx < 0 ? 0 : (kidx > HW - 1 ? HW - 1 : kidx);
      sKI[t] = kidx; sKMsk[t] = msk;
      sKX[t] = (float)(kidx / HH) + 0.5f;
      sKY[t] = (float)(kidx % HH) + 0.5f;
    }
    __syncthreads();
    // cross posembed hidden -> sH (f16)
    for (int e = t; e < 128 * 128; e += 256) {
      int k = e >> 7, c = e & 127;
      float s = A.pe_w1[c * 2] * sKX[k] + A.pe_w1[c * 2 + 1] * sKY[k] + A.pe_b1[c];
      float sc = A.pe_g[c] * rsqrtf(A.pe_v[c] + EPS);
      s = (s - A.pe_m[c]) * sc + A.pe_b[c];
      sH[k * 144 + c] = (_Float16)fmaxf(s, 0.f);
    }
    __syncthreads();

    v8f acc[8];
#pragma unroll
    for (int jn = 0; jn < 8; ++jn)
#pragma unroll
      for (int r = 0; r < 8; ++r) acc[jn][r] = 0.f;
    gemm128(sH, A.pew2, sB, acc, t);           // kpe
#pragma unroll
    for (int jn = 0; jn < 8; ++jn) {
      int n = jn * 16 + nl;
      float b2v = A.pe_b2[n];
#pragma unroll
      for (int r = 0; r < 8; ++r) {
        int m = wave * 16 + g * 8 + r;
        float kf = (float)A.feath[(size_t)(b * HW + sKI[m]) * HID + n];
        sKK[m * 144 + n] = (_Float16)(acc[jn][r] + b2v + kf);
      }
    }
    __syncthreads();

#pragma unroll
    for (int jn = 0; jn < 8; ++jn)
#pragma unroll
      for (int r = 0; r < 8; ++r) acc[jn][r] = 0.f;
    gemm128(sKK, A.wkh, sB, acc, t);           // Khat
#pragma unroll
    for (int jn = 0; jn < 8; ++jn) {
      int n = jn * 16 + nl;
      float bkv = A.bk[n];
#pragma unroll
      for (int r = 0; r < 8; ++r) {
        int m = wave * 16 + g * 8 + r;
        sH[m * 144 + n] = (_Float16)(acc[jn][r] + bkv);
      }
    }
    __syncthreads();
    { // scores: wave == head
      int hh = wave;
      for (int k = lane; k < 128; k += 32) {
        float s;
        if (sKMsk[k]) s = NEGINF;
        else {
          s = 0.f;
#pragma unroll
          for (int d = 0; d < 16; ++d) s += (float)sH[k * 144 + hh * 16 + d] * sQ[hh * 16 + d];
          s *= 0.25f;  // 1/sqrt(16)
        }
        sS[k * 8 + hh] = s;
      }
    }
    __syncthreads();

#pragma unroll
    for (int jn = 0; jn < 8; ++jn)
#pragma unroll
      for (int r = 0; r < 8; ++r) acc[jn][r] = 0.f;
    gemm128(sKK, A.wvh, sB, acc, t);           // V
#pragma unroll
    for (int jn = 0; jn < 8; ++jn) {
      int n = jn * 16 + nl;
      float bvv = A.bv[n];
#pragma unroll
      for (int r = 0; r < 8; ++r) {
        int m = wave * 16 + g * 8 + r;
        sH[m * 144 + n] = (_Float16)(acc[jn][r] + bvv);
      }
    }
    __syncthreads();
    if (t < 8) { // online softmax state per head
      int hh = t;
      float mo = sM[hh], tm = NEGINF;
      for (int k = 0; k < 128; ++k) tm = fmaxf(tm, sS[k * 8 + hh]);
      float mn = fmaxf(mo, tm);
      float al, ls;
      if (mn == NEGINF) {
        al = 1.f; ls = 0.f;
        for (int k = 0; k < 128; ++k) sS[k * 8 + hh] = 0.f;
      } else {
        al = (mo == NEGINF) ? 0.f : expf(mo - mn);
        ls = sL[hh] * al;
        for (int k = 0; k < 128; ++k) {
          float sk = sS[k * 8 + hh];
          float w = (sk == NEGINF) ? 0.f : expf(sk - mn);
          sS[k * 8 + hh] = w; ls += w;
        }
      }
      sM[hh] = mn; sL[hh] = ls; sAl[hh] = al;
    }
    __syncthreads();
    if (t < 128) {
      int c = t, hh = c >> 4;
      float o = sO[c] * sAl[hh];
      for (int k = 0; k < 128; ++k) o += sS[k * 8 + hh] * (float)sH[k * 144 + c];
      sO[c] = o;
    }
    __syncthreads();
  }

  if (t < 128) sO[t] = sO[t] / sL[t >> 4];
  __syncthreads();
  float pre2 = 0.f;
  if (t < 128) {
    float s = A.bo[t];
    for (int k = 0; k < 128; ++k) s += A.wo[t * 128 + k] * sO[k];
    pre2 = A.q1ws[(size_t)j * 128 + t] + s;
  }
  float q2 = ln256(pre2, t, sRed, A.n2g, A.n2b);
  if (t < 128) sVa[t] = q2;
  __syncthreads();
  float r1 = A.l1b[t];
  for (int k = 0; k < 128; ++k) r1 += A.l1w[t * 128 + k] * sVa[k];
  sVb[t] = fmaxf(r1, 0.f);
  __syncthreads();
  float pre3 = 0.f;
  if (t < 128) {
    float s = A.l2b[t];
    for (int k = 0; k < 256; ++k) s += A.l2w[t * 256 + k] * sVb[k];
    pre3 = q2 + s;
  }
  float q3 = ln256(pre3, t, sRed, A.n3g, A.n3b);
  if (t < 128) sVa[t] = q3;
  __syncthreads();
#pragma unroll
  for (int hI = 0; hI < 6; ++hI) {
    HeadP hp = A.heads[hI];
    if (t < 64) {
      float s = hp.b1[t];
      for (int k = 0; k < 128; ++k) s += hp.w1[t * 128 + k] * sVa[k];
      float sc = hp.g[t] * rsqrtf(hp.vv[t] + EPS);
      s = (s - hp.mm[t]) * sc + hp.be[t];
      sVb[t] = fmaxf(s, 0.f);
    }
    __syncthreads();
    if (t < hp.nout) {
      float s = hp.b2[t];
      for (int k = 0; k < 64; ++k) s += hp.w2[t * 64 + k] * sVb[k];
      A.preds[(size_t)(b * 20 + hp.cb + t) * NPROP + jj] = s;
    }
    __syncthreads();
  }
}

// ---------------------------------------------------------------- host launch
// d_in order: x, then params pytree with sorted dict keys at every level.
enum {
  I_X = 0, I_CE_B = 1, I_CE_W = 2,
  I_CA_BK = 3, I_CA_BO = 4, I_CA_BQ = 5, I_CA_BV = 6, I_CA_WK = 7, I_CA_WO = 8, I_CA_WQ = 9, I_CA_WV = 10,
  I_CPE_B1 = 11, I_CPE_B2 = 12, I_CPE_BNB = 13, I_CPE_BNG = 14, I_CPE_BNM = 15, I_CPE_BNV = 16, I_CPE_W1 = 17, I_CPE_W2 = 18,
  I_H_CENTER = 19, I_H_DIM = 27, I_H_HEATMAP = 35, I_H_HEIGHT = 43, I_H_ROT = 51, I_H_VEL = 59,
  I_HM1_B = 67, I_HM1_W = 68, I_HM2_B = 69, I_HM2_W = 70,
  I_HMBN_B = 71, I_HMBN_G = 72, I_HMBN_M = 73, I_HMBN_V = 74,
  I_L1B = 75, I_L1W = 76, I_L2B = 77, I_L2W = 78,
  I_N1B = 79, I_N1G = 80, I_N2B = 81, I_N2G = 82, I_N3B = 83, I_N3G = 84,
  I_SA_BK = 85, I_SA_BO = 86, I_SA_BQ = 87, I_SA_BV = 88, I_SA_WK = 89, I_SA_WO = 90, I_SA_WQ = 91, I_SA_WV = 92,
  I_SPE_B1 = 93, I_SPE_B2 = 94, I_SPE_BNB = 95, I_SPE_BNG = 96, I_SPE_BNM = 97, I_SPE_BNV = 98, I_SPE_W1 = 99, I_SPE_W2 = 100,
  I_SHARED_B = 101, I_SHARED_W = 102
};

extern "C" void kernel_launch(void* const* d_in, const int* in_sizes, int n_in,
                              void* d_out, int out_size, void* d_ws, size_t ws_size,
                              hipStream_t stream) {
  (void)in_sizes; (void)n_in; (void)out_size; (void)ws_size;
  auto P = [&](int i) { return (const float*)d_in[i]; };

  char* w = (char*)d_ws;
  auto take = [&](size_t bytes) { void* p = (void*)w; w += (bytes + 255) & ~(size_t)255; return p; };

  _Float16* XH    = (_Float16*)take((size_t)BB * HH * HH * CIN0 * 2);
  _Float16* WP1   = (_Float16*)take((size_t)128 * 9 * CIN0 * 2);
  _Float16* WP2   = (_Float16*)take((size_t)128 * 9 * HID * 2);
  _Float16* WP3   = (_Float16*)take((size_t)16 * 9 * HID * 2);
  _Float16* FEATH = (_Float16*)take((size_t)NPIX * HID * 2);
  _Float16* H2H   = (_Float16*)take((size_t)NPIX * HID * 2);
  float*    HSIG  = (float*)take((size_t)BB * NCLS * HW * 4);
  float*    HMASK = (float*)take((size_t)BB * NCLS * HW * 4);
  _Float16* PEW2H = (_Float16*)take(16384 * 2);
  _Float16* WKH   = (_Float16*)take(16384 * 2);
  _Float16* WVH   = (_Float16*)take(16384 * 2);
  int*      TOPCLS = (int*)take(NPR_T * 4);
  int*      TOPPIX = (int*)take(NPR_T * 4);
  float*    Q1WS  = (float*)take((size_t)NPR_T * 128 * 4);
  float*    QCWS  = (float*)take((size_t)NPR_T * 128 * 4);

  float* preds  = (float*)d_out;
  float* hm_out = (float*)d_out + BB * 20 * NPROP;  // dense_heatmap (B,10,180,180)

  // 1) transpose x NCHW->NHWC f16
  {
    int n = BB * CIN0 * HH * HH;
    k_convert_x<<<(n + 255) / 256, 256, 0, stream>>>(P(I_X), XH, n);
  }
  // 2) pack conv weights
  {
    int n1 = 128 * 9 * CIN0;
    k_pack_conv<<<(n1 + 255) / 256, 256, 0, stream>>>(P(I_SHARED_W), WP1, 128, CIN0, 128, n1);
    int n2 = 128 * 9 * HID;
    k_pack_conv<<<(n2 + 255) / 256, 256, 0, stream>>>(P(I_HM1_W), WP2, 128, HID, 128, n2);
    int n3 = 16 * 9 * HID;
    k_pack_conv<<<(n3 + 255) / 256, 256, 0, stream>>>(P(I_HM2_W), WP3, 10, HID, 16, n3);
  }
  // 3) f16 copies of attention/pe weight matrices
  k_half_copy<<<64, 256, 0, stream>>>(P(I_CPE_W2), PEW2H, 16384);
  k_half_copy<<<64, 256, 0, stream>>>(P(I_CA_WK), WKH, 16384);
  k_half_copy<<<64, 256, 0, stream>>>(P(I_CA_WV), WVH, 16384);

  const int convGrid = (NPIX + 127) / 128;  // 507
  // 4) shared conv (512->128), out = lidar/feat f16 NHWC
  k_conv<CIN0, 128, 0><<<convGrid, 256, 0, stream>>>(
      XH, WP1, P(I_SHARED_B), nullptr, nullptr, nullptr, nullptr, FEATH, nullptr, nullptr);
  // 5) heatmap conv1 (128->128) + BN + ReLU
  k_conv<HID, 128, 1><<<convGrid, 256, 0, stream>>>(
      FEATH, WP2, P(I_HM1_B), P(I_HMBN_G), P(I_HMBN_B), P(I_HMBN_M), P(I_HMBN_V),
      H2H, nullptr, nullptr);
  // 6) heatmap conv2 (128->10) -> dense_heatmap (d_out) + sigmoid map
  k_conv<HID, 16, 2><<<convGrid, 256, 0, stream>>>(
      H2H, WP3, P(I_HM2_B), nullptr, nullptr, nullptr, nullptr, nullptr, hm_out, HSIG);
  // 7) NMS
  {
    int n = BB * NCLS * HW;
    k_nms<<<(n + 255) / 256, 256, 0, stream>>>(HSIG, HMASK, n);
  }
  // 8) top-k
  k_topk<<<BB, 1024, 0, stream>>>(HMASK, TOPCLS, TOPPIX);
  // 9) per-proposal prep (query, self-attn, LN1, cross Q)
  k_prep<<<NPR_T, 128, 0, stream>>>(
      FEATH, P(I_CE_W), P(I_CE_B),
      P(I_SPE_W1), P(I_SPE_B1), P(I_SPE_BNG), P(I_SPE_BNB), P(I_SPE_BNM), P(I_SPE_BNV),
      P(I_SPE_W2), P(I_SPE_B2),
      P(I_SA_WV), P(I_SA_BV), P(I_SA_WO), P(I_SA_BO),
      P(I_N1G), P(I_N1B), P(I_CA_WQ), P(I_CA_BQ),
      TOPCLS, TOPPIX, Q1WS, QCWS);
  // 10) fused cross-attention decoder + FFN + heads
  {
    CrossArgs A;
    A.feath = FEATH; A.pew2 = PEW2H; A.wkh = WKH; A.wvh = WVH;
    A.pe_w1 = P(I_CPE_W1); A.pe_b1 = P(I_CPE_B1);
    A.pe_g = P(I_CPE_BNG); A.pe_b = P(I_CPE_BNB); A.pe_m = P(I_CPE_BNM); A.pe_v = P(I_CPE_BNV);
    A.pe_b2 = P(I_CPE_B2);
    A.bk = P(I_CA_BK); A.bv = P(I_CA_BV); A.wo = P(I_CA_WO); A.bo = P(I_CA_BO);
    A.n2g = P(I_N2G); A.n2b = P(I_N2B); A.n3g = P(I_N3G); A.n3b = P(I_N3B);
    A.l1w = P(I_L1W); A.l1b = P(I_L1B); A.l2w = P(I_L2W); A.l2b = P(I_L2B);
    A.toppix = TOPPIX; A.q1ws = Q1WS; A.qcws = QCWS;
    const int hbase[6] = {I_H_CENTER, I_H_HEIGHT, I_H_DIM, I_H_ROT, I_H_VEL, I_H_HEATMAP};
    const int hnout[6] = {2, 1, 3, 2, 2, 10};
    const int hcb[6]   = {0, 2, 3, 6, 8, 10};
    for (int i = 0; i < 6; ++i) {
      int base = hbase[i];
      A.heads[i].b1 = P(base + 0); A.heads[i].b2 = P(base + 1);
      A.heads[i].be = P(base + 2); A.heads[i].g  = P(base + 3);
      A.heads[i].mm = P(base + 4); A.heads[i].vv = P(base + 5);
      A.heads[i].w1 = P(base + 6); A.heads[i].w2 = P(base + 7);
      A.heads[i].nout = hnout[i]; A.heads[i].cb = hcb[i];
    }
    A.preds = preds;
    k_cross<<<NPR_T, 256, 0, stream>>>(A);
  }
}
